// BitLinear_27376121545313
// MI455X (gfx1250) — 8-bit matrix operand order repaired and checked by reading, compiled, not run on hardware
//
#include <hip/hip_runtime.h>

typedef __attribute__((ext_vector_type(8))) int v8i;

#define KDIM 4096
#define NDIM 11008
#define MDIM 16384
#define W_ELEMS (11008u * 4096u)
#define W_VEC4  (W_ELEMS / 4)

#define OFF_PART      0
#define OFF_GAMMA     4096
#define OFF_ROWSCALE  8192
#define OFF_WQ        81920
#define OFF_XQ        45170688

__global__ __launch_bounds__(256)
void k_wabs_partial(const float4* __restrict__ w, float* __restrict__ part) {
    __shared__ float sdata[256];
    float s = 0.0f;
    for (unsigned i = blockIdx.x * blockDim.x + threadIdx.x; i < W_VEC4;
         i += gridDim.x * blockDim.x) {
        float4 v = w[i];
        s += fabsf(v.x) + fabsf(v.y) + fabsf(v.z) + fabsf(v.w);
    }
    sdata[threadIdx.x] = s;
    __syncthreads();
    for (int st = 128; st > 0; st >>= 1) {
        if ((int)threadIdx.x < st) sdata[threadIdx.x] += sdata[threadIdx.x + st];
        __syncthreads();
    }
    if (threadIdx.x == 0) part[blockIdx.x] = sdata[0];
}

__global__ __launch_bounds__(256)
void k_gamma(const float* __restrict__ part, float* __restrict__ gamma) {
    __shared__ float sdata[256];
    float s = part[threadIdx.x] + part[threadIdx.x + 256] +
              part[threadIdx.x + 512] + part[threadIdx.x + 768];
    sdata[threadIdx.x] = s;
    __syncthreads();
    for (int st = 128; st > 0; st >>= 1) {
        if ((int)threadIdx.x < st) sdata[threadIdx.x] += sdata[threadIdx.x + st];
        __syncthreads();
    }
    if (threadIdx.x == 0) *gamma = sdata[0] / (float)W_ELEMS + 1e-5f;
}

__device__ __forceinline__ signed char tern_q(float v, float inv) {
    int q = (int)rintf(v * inv);
    q = q > 1 ? 1 : (q < -1 ? -1 : q);
    return (signed char)q;
}

__global__ __launch_bounds__(256)
void k_quant_w(const float4* __restrict__ w, const float* __restrict__ gammaPtr,
               char4* __restrict__ wq) {
    const float inv = 1.0f / *gammaPtr;
    for (unsigned i = blockIdx.x * blockDim.x + threadIdx.x; i < W_VEC4;
         i += gridDim.x * blockDim.x) {
        float4 v = w[i];
        char4 q;
        q.x = tern_q(v.x, inv);
        q.y = tern_q(v.y, inv);
        q.z = tern_q(v.z, inv);
        q.w = tern_q(v.w, inv);
        wq[i] = q;
    }
}

__device__ __forceinline__ signed char act_q(float v, float s) {
    int q = (int)rintf(v * s);
    q = q > 127 ? 127 : (q < -128 ? -128 : q);
    return (signed char)q;
}

__global__ __launch_bounds__(256)
void k_quant_x(const float4* __restrict__ x, const float* __restrict__ gwPtr,
               char4* __restrict__ xq, float* __restrict__ rowscale) {
    __shared__ float sdata[256];
    const unsigned row = blockIdx.x;
    const float4* xr = x + (size_t)row * (KDIM / 4);
    char4* qr = xq + (size_t)row * (KDIM / 4);

    float4 v[4];
    float mx = 0.0f;
#pragma unroll
    for (int i = 0; i < 4; ++i) {
        v[i] = xr[threadIdx.x + 256 * i];
        mx = fmaxf(mx, fmaxf(fmaxf(fabsf(v[i].x), fabsf(v[i].y)),
                             fmaxf(fabsf(v[i].z), fabsf(v[i].w))));
    }
    sdata[threadIdx.x] = mx;
    __syncthreads();
    for (int st = 128; st > 0; st >>= 1) {
        if ((int)threadIdx.x < st)
            sdata[threadIdx.x] = fmaxf(sdata[threadIdx.x], sdata[threadIdx.x + st]);
        __syncthreads();
    }
    const float gamma = sdata[0] + 1e-5f;
    if (threadIdx.x == 0)
        rowscale[row] = gamma * (*gwPtr) * (1.0f / 127.0f);
    const float s = 127.0f / gamma;
#pragma unroll
    for (int i = 0; i < 4; ++i) {
        char4 q;
        q.x = act_q(v[i].x, s);
        q.y = act_q(v[i].y, s);
        q.z = act_q(v[i].z, s);
        q.w = act_q(v[i].w, s);
        qr[threadIdx.x + 256 * i] = q;
    }
}

__global__ __launch_bounds__(256)
void k_gemm_iu8(const signed char* __restrict__ xq,
                const signed char* __restrict__ wq,
                const float* __restrict__ rowscale,
                float* __restrict__ out) {
    const int lane = threadIdx.x & 31;
    const int wave = threadIdx.x >> 5;
    const int g    = lane >> 4;
    const int l16  = lane & 15;

    const int wm = wave >> 2;
    const int wn = wave & 3;
    const int mBase = blockIdx.y * 128 + wm * 64;
    const int nBase = blockIdx.x * 128 + wn * 32;

    v8i acc[4][2] = {};

    const signed char* arow[4];
#pragma unroll
    for (int i = 0; i < 4; ++i)
        arow[i] = xq + (size_t)(mBase + 16 * i + l16) * KDIM;
    const signed char* brow[2];
#pragma unroll
    for (int j = 0; j < 2; ++j)
        brow[j] = wq + (size_t)(nBase + 16 * j + l16) * KDIM;

    for (int k0 = 0; k0 < KDIM; k0 += 64) {
        __builtin_prefetch(arow[0] + k0 + 256, 0, 3);
        __builtin_prefetch(brow[0] + k0 + 256, 0, 3);

        v8i a[4];
#pragma unroll
        for (int i = 0; i < 4; ++i) {
            const signed char* p = arow[i] + k0 + 16 * g;
#pragma unroll
            for (int jj = 0; jj < 4; ++jj) {
                int2 d = *(const int2*)(p + 8 * (jj & 1) + 32 * (jj >> 1));
                a[i][2 * jj]     = d.x;
                a[i][2 * jj + 1] = d.y;
            }
        }
        v8i b[2];
#pragma unroll
        for (int j = 0; j < 2; ++j) {
            const signed char* p = brow[j] + k0 + 16 * g;
            int4 d0 = *(const int4*)(p);
            int4 d1 = *(const int4*)(p + 32);
            b[j][0] = d0.x; b[j][1] = d0.y; b[j][2] = d0.z; b[j][3] = d0.w;
            b[j][4] = d1.x; b[j][5] = d1.y; b[j][6] = d1.z; b[j][7] = d1.w;
        }
#pragma unroll
        for (int i = 0; i < 4; ++i)
#pragma unroll
            for (int j = 0; j < 2; ++j)
                acc[i][j] = __builtin_amdgcn_wmma_i32_16x16x64_iu8(
                    true, a[i], true, b[j], acc[i][j],
                     false,  false);
    }

#pragma unroll
    for (int i = 0; i < 4; ++i) {
        const int mrowBase = mBase + 16 * i + 8 * g;
#pragma unroll
        for (int r = 0; r < 8; ++r) {
            const int m = mrowBase + r;
            const float s = rowscale[m];
#pragma unroll
            for (int j = 0; j < 2; ++j) {
                const int n = nBase + 16 * j + l16;
                out[(size_t)m * NDIM + n] = (float)acc[i][j][r] * s;
            }
        }
    }
}

extern "C" void kernel_launch(void* const* d_in, const int* in_sizes, int n_in,
                              void* d_out, int out_size, void* d_ws, size_t ws_size,
                              hipStream_t stream) {
    const float* x = (const float*)d_in[0];
    const float* w = (const float*)d_in[1];
    float* out = (float*)d_out;
    char* ws = (char*)d_ws;

    float*       part     = (float*)(ws + OFF_PART);
    float*       gamma    = (float*)(ws + OFF_GAMMA);
    float*       rowscale = (float*)(ws + OFF_ROWSCALE);
    signed char* wq       = (signed char*)(ws + OFF_WQ);
    signed char* xq       = (signed char*)(ws + OFF_XQ);

    k_wabs_partial<<<1024, 256, 0, stream>>>((const float4*)w, part);
    k_gamma<<<1, 256, 0, stream>>>(part, gamma);
    k_quant_w<<<2048, 256, 0, stream>>>((const float4*)w, gamma, (char4*)wq);
    k_quant_x<<<MDIM, 256, 0, stream>>>((const float4*)x, gamma, (char4*)xq, rowscale);

    dim3 grid(NDIM / 128, MDIM / 128);
    k_gemm_iu8<<<grid, 256, 0, stream>>>(xq, wq, rowscale, out);
}
